// SparseMultiAttention_7121055777172
// MI455X (gfx1250) — compile-verified, matrix operand order checked by reading, not run on hardware
//
#include <hip/hip_runtime.h>
#include <hip/hip_bf16.h>
#include <math.h>

// ---------------------------------------------------------------------------
// GAT-style sparse multi-attention for MI455X (gfx1250, wave32).
//   K1: Wh = h @ W via v_wmma_f32_16x16x32_f16 with fp16-split (hi/lo) operands
//       -> fp32-class accuracy while exercising the CDNA5 matrix pipe.
//   K2: per-node projections s = Wh . a  (4 dot products per node)
//   K3: per-edge attention logits + global max (wave32 shuffle + atomicMax on
//       positive-float bits)
//   K4: exp(v - max) in place + global sum (atomicAdd)
//   K5: gather-aggregate h' = (1/Z) * sum_k val * Wh[dst], then ELU.
// Everything is HBM/L2-bandwidth bound (Wh = 12.8 MB lives in the 192 MB L2),
// so the fp16-split tripling of WMMA count is free.
// ---------------------------------------------------------------------------

typedef __attribute__((ext_vector_type(16))) _Float16 v16h;
typedef __attribute__((ext_vector_type(8)))  float    v8f;

#define IN_F  128
#define OUT_F 64
#define DEG   16
#define ALPHA 0.2f

// ---------------- K1: WMMA GEMM  Wh[N x 64] = h[N x 128] @ W[128 x 64] ------
__global__ void gat_gemm_wh(const float* __restrict__ h,
                            const float* __restrict__ W,
                            float* __restrict__ Wh, int N) {
  const int wave = (blockIdx.x * blockDim.x + threadIdx.x) >> 5;   // row tile
  const int lane = threadIdx.x & 31;
  const int row0 = wave * 16;
  if (row0 >= N) return;                       // wave-uniform guard (EXEC all-1)

  const int m    = lane & 15;
  const int koff = (lane & 16) ? 8 : 0;        // K sub-offset per half-wave
  const int roff = (lane & 16) ? 8 : 0;        // C/D row sub-offset

  v8f acc[4] = {v8f{}, v8f{}, v8f{}, v8f{}};

  #pragma unroll
  for (int ks = 0; ks < 4; ++ks) {             // K = 128 in steps of 32
    const int kbase = ks * 32;

    // ---- A fragment (row-major h), fp16 hi/lo split ----
    v16h a_hi, a_lo;
    const float* hrow = h + (size_t)(row0 + m) * IN_F + kbase + koff;
    #pragma unroll
    for (int t = 0; t < 8; ++t) {
      float x = hrow[t];                       // K = kbase + koff + t
      float y = hrow[16 + t];                  // K = kbase + 16 + koff + t
      _Float16 xh = (_Float16)x;
      _Float16 yh = (_Float16)y;
      a_hi[t]     = xh;  a_lo[t]     = (_Float16)(x - (float)xh);
      a_hi[8 + t] = yh;  a_lo[8 + t] = (_Float16)(y - (float)yh);
    }

    #pragma unroll
    for (int ct = 0; ct < 4; ++ct) {           // 4 col tiles of 16 -> 64 cols
      const int n0 = ct * 16;
      // ---- B fragment (row-major W: W[k*64 + n]), fp16 hi/lo split ----
      v16h b_hi, b_lo;
      const float* wcol = W + (size_t)(kbase + koff) * OUT_F + n0 + m;
      #pragma unroll
      for (int t = 0; t < 8; ++t) {
        float x = wcol[(size_t)t * OUT_F];         // K = kbase + koff + t
        float y = wcol[(size_t)(16 + t) * OUT_F];  // K = kbase + 16 + koff + t
        _Float16 xh = (_Float16)x;
        _Float16 yh = (_Float16)y;
        b_hi[t]     = xh;  b_lo[t]     = (_Float16)(x - (float)xh);
        b_hi[8 + t] = yh;  b_lo[8 + t] = (_Float16)(y - (float)yh);
      }
      // acc += Ahi*Bhi + Ahi*Blo + Alo*Bhi  (Alo*Blo ~ 2^-22, dropped)
      acc[ct] = __builtin_amdgcn_wmma_f32_16x16x32_f16(
          false, a_hi, false, b_hi, (short)0, acc[ct], false, false);
      acc[ct] = __builtin_amdgcn_wmma_f32_16x16x32_f16(
          false, a_hi, false, b_lo, (short)0, acc[ct], false, false);
      acc[ct] = __builtin_amdgcn_wmma_f32_16x16x32_f16(
          false, a_lo, false, b_hi, (short)0, acc[ct], false, false);
    }
  }

  // ---- store C: v8f elem r -> row (row0 + roff + r), col (n0 + m) ----
  #pragma unroll
  for (int ct = 0; ct < 4; ++ct) {
    #pragma unroll
    for (int r = 0; r < 8; ++r) {
      Wh[(size_t)(row0 + roff + r) * OUT_F + ct * 16 + m] = acc[ct][r];
    }
  }
}

// ---------------- K2: per-node projections s_j[i] = Wh[i,:] . a_j -----------
__global__ void gat_proj(const float* __restrict__ Wh,
                         const float* __restrict__ a,
                         float* __restrict__ s, int N) {
  __shared__ float sa[4 * OUT_F];
  if (threadIdx.x < 4 * OUT_F) sa[threadIdx.x] = a[threadIdx.x];
  __syncthreads();
  const int i = blockIdx.x * blockDim.x + threadIdx.x;
  if (i >= N) return;
  const float* row = Wh + (size_t)i * OUT_F;
  float s0 = 0.f, s1 = 0.f, s2 = 0.f, s3 = 0.f;
  #pragma unroll
  for (int c = 0; c < OUT_F; ++c) {
    const float v = row[c];
    s0 += v * sa[c];
    s1 += v * sa[OUT_F + c];
    s2 += v * sa[2 * OUT_F + c];
    s3 += v * sa[3 * OUT_F + c];
  }
  s[i]         = s0;   // a1, network 0
  s[N + i]     = s1;   // a2, network 0
  s[2 * N + i] = s2;   // a1, network 1
  s[3 * N + i] = s3;   // a2, network 1
}

// ---------------- K5-init: reduction scalars --------------------------------
__global__ void gat_init(int* gmax_bits, float* gsum) {
  *gmax_bits = 0;      // vals are strictly positive -> int-bit max works, 0 ok
  *gsum = 0.f;
}

// ---------------- K3: edge logits + global max ------------------------------
__global__ void gat_edge_vals(const int* __restrict__ ei,
                              const float* __restrict__ s,
                              const float* __restrict__ nw,
                              float* __restrict__ vals,
                              int* __restrict__ gmax_bits, int N, int E) {
  const int e = blockIdx.x * blockDim.x + threadIdx.x;
  float v = 0.f;
  if (e < E) {
    const int i = ei[e];        // src
    const int j = ei[E + e];    // dst
    // wts = softmax(nw); the reference uses wts[0,0] for BOTH networks.
    const float wts00 = 1.f / (1.f + expf(nw[1] - nw[0]));
    const float e0 = s[i]         + s[N + j];
    const float e1 = s[2 * N + i] + s[3 * N + j];
    const float l0 = (e0 > 0.f) ? e0 : ALPHA * e0;
    const float l1 = (e1 > 0.f) ? e1 : ALPHA * e1;
    v = wts00 * (expf(-l0) + expf(-l1));       // > 0 always
    vals[e] = v;
  }
  // wave32 max reduction, then block, then global atomic on positive bits
  #pragma unroll
  for (int off = 16; off > 0; off >>= 1)
    v = fmaxf(v, __shfl_down(v, off, 32));
  __shared__ float red[8];
  const int lane = threadIdx.x & 31, wid = threadIdx.x >> 5;
  if (lane == 0) red[wid] = v;
  __syncthreads();
  if (threadIdx.x == 0) {
    float m = red[0];
    #pragma unroll
    for (int w = 1; w < 8; ++w) m = fmaxf(m, red[w]);
    atomicMax(gmax_bits, __float_as_int(m));
  }
}

// ---------------- K4: exp(v - max) in place + global sum --------------------
__global__ void gat_expsum(float* __restrict__ vals,
                           const int* __restrict__ gmax_bits,
                           float* __restrict__ gsum, int E) {
  const int e = blockIdx.x * blockDim.x + threadIdx.x;
  const float vmax = __int_as_float(*gmax_bits);
  float ev = 0.f;
  if (e < E) {
    ev = expf(vals[e] - vmax);
    vals[e] = ev;
  }
  #pragma unroll
  for (int off = 16; off > 0; off >>= 1)
    ev += __shfl_down(ev, off, 32);
  __shared__ float red[8];
  const int lane = threadIdx.x & 31, wid = threadIdx.x >> 5;
  if (lane == 0) red[wid] = ev;
  __syncthreads();
  if (threadIdx.x == 0) {
    float t = 0.f;
    #pragma unroll
    for (int w = 0; w < 8; ++w) t += red[w];
    atomicAdd(gsum, t);
  }
}

// ---------------- K5: aggregate + ELU ---------------------------------------
// One thread per (node, col); 64 consecutive threads share dst -> coalesced
// Wh reads that stay resident in the 192 MB L2.
__global__ void gat_aggregate(const float* __restrict__ Wh,
                              const float* __restrict__ vals,
                              const int* __restrict__ ei,
                              const float* __restrict__ gsum,
                              float* __restrict__ out, int N, int E) {
  const int idx = blockIdx.x * blockDim.x + threadIdx.x;
  if (idx >= N * OUT_F) return;
  const int node = idx >> 6;
  const int col  = idx & 63;
  const float inv = 1.f / (*gsum);
  float acc = 0.f;
  #pragma unroll
  for (int k = 0; k < DEG; ++k) {
    const int e = node * DEG + k;
    const int j = ei[E + e];
    acc += vals[e] * Wh[(size_t)j * OUT_F + col];
  }
  acc *= inv;
  out[idx] = (acc > 0.f) ? acc : expm1f(acc);   // ELU(alpha=1)
}

// ---------------------------------------------------------------------------
extern "C" void kernel_launch(void* const* d_in, const int* in_sizes, int n_in,
                              void* d_out, int out_size, void* d_ws, size_t ws_size,
                              hipStream_t stream) {
  const float* h  = (const float*)d_in[0];   // (N, 128)
  const int*   ei = (const int*)  d_in[1];   // (2, E)
  const float* W  = (const float*)d_in[2];   // (128, 64)
  const float* a  = (const float*)d_in[3];   // (256, 1)
  const float* nw = (const float*)d_in[4];   // (2, 1)

  const int N = in_sizes[0] / IN_F;          // 50000
  const int E = in_sizes[1] / 2;             // 800000

  // workspace layout (floats): Wh | s(4N) | vals(E) | [gmax_bits, gsum]
  float* ws    = (float*)d_ws;
  float* Wh    = ws;
  float* s     = Wh + (size_t)N * OUT_F;
  float* vals  = s + (size_t)4 * N;
  float* scal  = vals + (size_t)E;
  int*   gmaxb = (int*)scal;
  float* gsum  = scal + 1;

  float* out = (float*)d_out;

  gat_init<<<1, 1, 0, stream>>>(gmaxb, gsum);

  const int rowTiles = (N + 15) / 16;                       // 3125
  gat_gemm_wh<<<(rowTiles + 7) / 8, 256, 0, stream>>>(h, W, Wh, N);

  gat_proj<<<(N + 255) / 256, 256, 0, stream>>>(Wh, a, s, N);

  gat_edge_vals<<<(E + 255) / 256, 256, 0, stream>>>(ei, s, nw, vals, gmaxb, N, E);

  gat_expsum<<<(E + 255) / 256, 256, 0, stream>>>(vals, gmaxb, gsum, E);

  gat_aggregate<<<((size_t)N * OUT_F + 255) / 256, 256, 0, stream>>>(
      Wh, vals, ei, gsum, out, N, E);
}
